// EntityLinker_87471303950751
// MI455X (gfx1250) — matrix operand order repaired and checked by reading, not compiled, not run on hardware
//
#include <hip/hip_runtime.h>

// ---------------------------------------------------------------------------
// EntityLinker edge-MLP for MI455X (gfx1250, wave32, WMMA)
//
//   h_i = node[src], h_j = node[dst]
//   feat = [h_i, h_j, |h_i-h_j|, h_i*h_j]            [E, 512]
//   x1 = relu(feat @ W1 + b1)                        [E, 256]
//   x2 = relu(x1 @ W2 + b2)                          [E, 128]
//   out = x2 @ W3 + b3                               [E, 2]
//
// Compute-bound (~164 GFLOP vs ~35 MB HBM traffic) -> f16 WMMA, f32 accum.
// This revision adds depth-4 software pipelining of B-fragment loads and
// up-front LDS A-fragment preloads so the WMMA pipe never serializes on a
// single in-flight load pair.
// ---------------------------------------------------------------------------

typedef _Float16 v16h __attribute__((ext_vector_type(16)));
typedef _Float16 v8h  __attribute__((ext_vector_type(8)));
typedef float    v8f  __attribute__((ext_vector_type(8)));
typedef short    v8s  __attribute__((ext_vector_type(8)));

#define N_NODES 50000
#define N_EDGES 500000
#define H       128

// workspace byte offsets (all 32B aligned)
#define OFF_NODE16 0u
#define OFF_W1T    12800000u   // 50000*128*2
#define OFF_W2T    13062144u   // + 256*512*2
#define OFF_W3T    13127680u   // + 128*256*2
// total ws needed: 13,131,776 bytes

// padded LDS strides (f16 units): row stride = 4 dwords mod 64 banks -> the
// 16 lanes of an A-fragment ds_load_b128 land on disjoint bank groups.
#define FSTRIDE 136   // 128 + 8
#define XSTRIDE 264   // 256 + 8
#define WAVE_LDS (16*FSTRIDE + 16*XSTRIDE)   // 6400 f16 = 12.8 KB per wave

static __device__ __forceinline__ v16h cat16(v8h lo, v8h hi) {
  return __builtin_shufflevector(lo, hi, 0,1,2,3,4,5,6,7,8,9,10,11,12,13,14,15);
}
static __device__ __forceinline__ v8h habs8(v8h x) {
  v8s u = __builtin_bit_cast(v8s, x);
  u = u & (short)0x7FFF;
  return __builtin_bit_cast(v8h, u);
}
static __device__ __forceinline__ v8f wmma_f16(v16h a, v16h b, v8f c) {
  // v_wmma_f32_16x16x32_f16  (neg_a, A, neg_b, B, c_mod, C, reuse_a, reuse_b)
  return __builtin_amdgcn_wmma_f32_16x16x32_f16(false, a, false, b, (short)0, c,
                                                false, false);
}

// ---------------------------------------------------------------------------
// Prep: f32->f16 node table; weights transposed to n-major f16 so each WMMA
// B-fragment is a lane-contiguous 32B read.  W3 padded to N=16 with zeros.
// ---------------------------------------------------------------------------
__global__ void prep_kernel(const float* __restrict__ node,
                            const float* __restrict__ W1,
                            const float* __restrict__ W2,
                            const float* __restrict__ W3,
                            _Float16* __restrict__ node16,
                            _Float16* __restrict__ w1t,
                            _Float16* __restrict__ w2t,
                            _Float16* __restrict__ w3t)
{
  size_t id = (size_t)blockIdx.x * blockDim.x + threadIdx.x;
  if (id < (size_t)N_NODES * H) { node16[id] = (_Float16)node[id]; return; }
  id -= (size_t)N_NODES * H;
  if (id < 256u * 512u) {                 // W1T[n][k] = W1[k][n], n<256, k<512
    size_t n = id >> 9, k = id & 511;
    w1t[id] = (_Float16)W1[k * 256 + n];
    return;
  }
  id -= 256u * 512u;
  if (id < 128u * 256u) {                 // W2T[n][k] = W2[k][n], n<128, k<256
    size_t n = id >> 8, k = id & 255;
    w2t[id] = (_Float16)W2[k * 128 + n];
    return;
  }
  id -= 128u * 256u;
  if (id < 16u * 128u) {                  // W3T[n][k], N padded 2 -> 16
    size_t n = id >> 7, k = id & 127;
    w3t[id] = (n < 2) ? (_Float16)W3[k * 2 + n] : (_Float16)0.0f;
  }
}

// ---------------------------------------------------------------------------
// Main: 4 waves/block, each wave owns a 16-edge tile end-to-end.
// No cross-wave sharing -> no s_barrier; LDS is in-order within a wave.
// ---------------------------------------------------------------------------
__global__ __launch_bounds__(128) void edge_mlp_kernel(
    const int* __restrict__ src, const int* __restrict__ dst,
    const _Float16* __restrict__ node16,
    const _Float16* __restrict__ w1t, const float* __restrict__ b1,
    const _Float16* __restrict__ w2t, const float* __restrict__ b2,
    const _Float16* __restrict__ w3t, const float* __restrict__ b3,
    float* __restrict__ out)
{
  __shared__ _Float16 lds[4 * WAVE_LDS];   // 51.2 KB

  const int lane = threadIdx.x & 31;
  const int wave = threadIdx.x >> 5;
  const int e  = lane & 15;     // A-fragment row / B-fragment column-in-tile
  const int hf = lane >> 4;     // lane half selects K sub-chunk
  const long long tileBase = (long long)blockIdx.x * 64 + wave * 16;
  if (tileBase >= N_EDGES) return;   // wave-uniform; E % 16 == 0 so tiles full

  _Float16* feat = lds + wave * WAVE_LDS;  // [16][FSTRIDE] feature / x2 stage
  _Float16* x1s  = feat + 16 * FSTRIDE;    // [16][XSTRIDE] x1 stage

  const int si = src[tileBase + e];
  const int di = dst[tileBase + e];
  const v8h* __restrict__ pi = (const v8h*)(node16 + (size_t)si * H + hf * 64);
  const v8h* __restrict__ pj = (const v8h*)(node16 + (size_t)di * H + hf * 64);

  // ---- Layer 1: [16,512] x [512,256], K chunked in 4 feature blocks -------
  v8f acc1[16] = {};
  #pragma unroll
  for (int c = 0; c < 4; ++c) {
    // lane (e,hf) produces feat[e][hf*64 .. hf*64+63] for this chunk
    v8h* fp = (v8h*)(feat + e * FSTRIDE + hf * 64);
    #pragma unroll
    for (int t = 0; t < 8; ++t) {
      v8h v;
      if      (c == 0) v = pi[t];
      else if (c == 1) v = pj[t];
      else if (c == 2) v = habs8(pi[t] - pj[t]);
      else             v = pi[t] * pj[t];
      fp[t] = v;
    }
    __builtin_amdgcn_wave_barrier();   // keep DS store -> cross-lane load order

    // Preload the chunk's 4 A fragments from LDS (no ds deps in the hot loop)
    v16h afr[4];
    #pragma unroll
    for (int kt = 0; kt < 4; ++kt) {
      const _Float16* ap = feat + e * FSTRIDE + kt * 32 + hf * 16;
      afr[kt] = cat16(*(const v8h*)ap, *(const v8h*)(ap + 8));
    }

    // Depth-4 rolling prefetch of B fragments over flattened (kt, nt)
    const _Float16* w1base = w1t + (size_t)e * 512 + (size_t)(c * 128 + hf * 16);
    v16h bq[4];
    #pragma unroll
    for (int p = 0; p < 4; ++p)
      bq[p] = *(const v16h*)(w1base + (size_t)p * (16 * 512));
    #pragma unroll
    for (int i = 0; i < 64; ++i) {
      const int kt = i >> 4, nt = i & 15;
      v16h bc = bq[i & 3];
      const int j = i + 4;
      if (j < 64)
        bq[i & 3] = *(const v16h*)(w1base + (size_t)(j & 15) * (16 * 512)
                                          + (size_t)(j >> 4) * 32);
      acc1[nt] = wmma_f16(afr[kt], bc, acc1[nt]);
    }
    __builtin_amdgcn_wave_barrier();   // chunk buffer reused next iteration
  }

  // ---- x1 = relu(acc1 + b1) -> LDS f16, row-major [m][n] ------------------
  #pragma unroll
  for (int nt = 0; nt < 16; ++nt) {
    const int n = nt * 16 + e;
    const float bias = b1[n];
    #pragma unroll
    for (int v = 0; v < 8; ++v) {
      const int m = v + hf * 8;             // C layout: lanes 16-31 hold M=8..15
      float val = acc1[nt][v] + bias;
      x1s[m * XSTRIDE + n] = (_Float16)(val > 0.f ? val : 0.f);
    }
  }
  __builtin_amdgcn_wave_barrier();

  // ---- Layer 2: [16,256] x [256,128] --------------------------------------
  v8f acc2[8] = {};
  v16h a2[8];                               // all 8 A fragments up front
  #pragma unroll
  for (int kt = 0; kt < 8; ++kt) {
    const _Float16* ap = x1s + e * XSTRIDE + kt * 32 + hf * 16;
    a2[kt] = cat16(*(const v8h*)ap, *(const v8h*)(ap + 8));
  }
  {
    const _Float16* w2base = w2t + (size_t)e * 256 + (size_t)(hf * 16);
    v16h bq[4];
    #pragma unroll
    for (int p = 0; p < 4; ++p)
      bq[p] = *(const v16h*)(w2base + (size_t)p * (16 * 256));
    #pragma unroll
    for (int i = 0; i < 64; ++i) {
      const int kt = i >> 3, nt = i & 7;
      v16h bc = bq[i & 3];
      const int j = i + 4;
      if (j < 64)
        bq[i & 3] = *(const v16h*)(w2base + (size_t)(j & 7) * (16 * 256)
                                          + (size_t)(j >> 3) * 32);
      acc2[nt] = wmma_f16(a2[kt], bc, acc2[nt]);
    }
  }

  // ---- x2 = relu(acc2 + b2) -> reuse feat buffer --------------------------
  #pragma unroll
  for (int nt = 0; nt < 8; ++nt) {
    const int n = nt * 16 + e;
    const float bias = b2[n];
    #pragma unroll
    for (int v = 0; v < 8; ++v) {
      const int m = v + hf * 8;
      float val = acc2[nt][v] + bias;
      feat[m * FSTRIDE + n] = (_Float16)(val > 0.f ? val : 0.f);
    }
  }
  __builtin_amdgcn_wave_barrier();

  // ---- Layer 3: [16,128] x [128,16] (N padded; only n<2 meaningful) -------
  v8f acc3 = {};
  {
    v16h a3[4];
    v16h b3q[4];
    #pragma unroll
    for (int kt = 0; kt < 4; ++kt) {
      const _Float16* ap = feat + e * FSTRIDE + kt * 32 + hf * 16;
      a3[kt]  = cat16(*(const v8h*)ap, *(const v8h*)(ap + 8));
      b3q[kt] = *(const v16h*)(w3t + (size_t)e * 128 + kt * 32 + hf * 16);
    }
    #pragma unroll
    for (int kt = 0; kt < 4; ++kt)
      acc3 = wmma_f16(a3[kt], b3q[kt], acc3);
  }

  if (e < 2) {                       // lanes 0,1,16,17 hold the C=2 outputs
    const float bias = b3[e];
    #pragma unroll
    for (int v = 0; v < 8; ++v) {
      const int m = v + hf * 8;
      out[(tileBase + m) * 2 + e] = acc3[v] + bias;
    }
  }
}

// ---------------------------------------------------------------------------
extern "C" void kernel_launch(void* const* d_in, const int* in_sizes, int n_in,
                              void* d_out, int out_size, void* d_ws, size_t ws_size,
                              hipStream_t stream)
{
  (void)in_sizes; (void)n_in; (void)out_size; (void)ws_size;

  const float* node = (const float*)d_in[0];
  const int*   src  = (const int*)  d_in[1];
  const int*   dst  = (const int*)  d_in[2];
  const float* W1   = (const float*)d_in[3];
  const float* b1   = (const float*)d_in[4];
  const float* W2   = (const float*)d_in[5];
  const float* b2   = (const float*)d_in[6];
  const float* W3   = (const float*)d_in[7];
  const float* b3   = (const float*)d_in[8];

  char* ws = (char*)d_ws;
  _Float16* node16 = (_Float16*)(ws + OFF_NODE16);
  _Float16* w1t    = (_Float16*)(ws + OFF_W1T);
  _Float16* w2t    = (_Float16*)(ws + OFF_W2T);
  _Float16* w3t    = (_Float16*)(ws + OFF_W3T);

  // prep: 6,400,000 + 131,072 + 32,768 + 2,048 = 6,565,888 elements
  const size_t prep_total = (size_t)N_NODES * H + 256 * 512 + 128 * 256 + 16 * 128;
  const int prep_blocks = (int)((prep_total + 255) / 256);
  prep_kernel<<<prep_blocks, 256, 0, stream>>>(node, W1, W2, W3,
                                               node16, w1t, w2t, w3t);

  const int blocks = (N_EDGES + 63) / 64;   // 64 edges (4 wave-tiles) / block
  edge_mlp_kernel<<<blocks, 128, 0, stream>>>(src, dst, node16,
                                              w1t, b1, w2t, b2, w3t, b3,
                                              (float*)d_out);
}
